// Retriever_42331197669532
// MI455X (gfx1250) — matrix operand order repaired and checked by reading, not compiled, not run on hardware
//
#include <hip/hip_runtime.h>
#include <hip/hip_bf16.h>
#include <stdint.h>

// ---------------- problem constants ----------------
#define BROWS   4096
#define DDIM    768
#define MEVID   100000
#define TOPK    10

#define NCHUNK  10            // E conversion chunks
#define CHUNK   10000         // E rows per chunk (10 * 10000 = 100000 exact)
#define NSEG    4             // col segments per chunk (per gemm block)
#define SEGCOLS 2500          // CHUNK / NSEG
#define NSEGPAIRS 79          // ceil(ceil(2500/16)/2) tile-pairs (32 cols each)
#define NCANDSEG (NCHUNK * NSEG)   // 40 candidate lists per row

typedef __attribute__((ext_vector_type(16))) __bf16 v16bf;
typedef __attribute__((ext_vector_type(8)))  float  v8f;

union Frag {
    v16bf v;
    uint4 q[2];
};

// round-to-nearest-even f32 -> bf16 (bit pattern)
__device__ __forceinline__ unsigned f2bf_bits(float f) {
    union { float f; unsigned u; } a; a.f = f;
    unsigned u = a.u;
    return (u + 0x7FFFu + ((u >> 16) & 1u)) >> 16;
}

// sorted-descending top-10 insertion; fully unrolled, registers only
__device__ __forceinline__ void insert10(float (&s)[TOPK], int (&id)[TOPK],
                                         float v, int c) {
    if (v <= s[TOPK - 1]) return;
    s[TOPK - 1] = v; id[TOPK - 1] = c;
#pragma unroll
    for (int j = TOPK - 1; j > 0; --j) {
        if (s[j] > s[j - 1]) {
            float ts = s[j]; s[j] = s[j - 1]; s[j - 1] = ts;
            int   ti = id[j]; id[j] = id[j - 1]; id[j - 1] = ti;
        }
    }
}

// ---------------- kernel 0: f32 -> bf16 conversion ----------------
__global__ void cvt_bf16_kernel(const float* __restrict__ src,
                                unsigned short* __restrict__ dst, int npairs) {
    int i = blockIdx.x * blockDim.x + threadIdx.x;
    if (i >= npairs) return;
    float2 v = ((const float2*)src)[i];
    unsigned lo = f2bf_bits(v.x), hi = f2bf_bits(v.y);
    ((unsigned*)dst)[i] = lo | (hi << 16);
}

// ---------------- kernel 1: fused bf16 WMMA GEMM + per-segment top-10 -----
// Each wave computes a 16x32 output slab (two 16x16 WMMA tiles sharing the
// A fragment), so per k-step: 2 ds_load_b128 (A) + 4 global_load_b128 (B)
// feed 2 WMMAs -> half the LDS traffic per WMMA vs a single-tile wave.
__global__ __launch_bounds__(256)
void gemm_topk_kernel(const unsigned short* __restrict__ Xbf,
                      const unsigned short* __restrict__ Ebf,
                      float2* __restrict__ cand, int chunk) {
    __shared__ __attribute__((aligned(16))) unsigned short As[16][776]; // +8 pad
    __shared__ float tile[8][16][33];      // per-wave 16x32 D slab (+pad)
    __shared__ float mrg_s[16][16][TOPK];  // [row][list][k]
    __shared__ int   mrg_i[16][16][TOPK];

    const int tid  = threadIdx.x;
    const int wave = tid >> 5;
    const int lane = tid & 31;
    const int ln15 = lane & 15;
    const int half = lane >> 4;
    const int rt   = blockIdx.x;   // row tile (16 rows of x)
    const int seg  = blockIdx.y;   // column segment within chunk

    // stage 16x768 bf16 x-tile into LDS
    {
        const unsigned* xsrc = (const unsigned*)(Xbf + (size_t)rt * 16 * DDIM);
        for (int i = tid; i < 16 * DDIM / 2; i += 256) {
            int e = i * 2;
            int r = e / DDIM, k = e % DDIM;
            *(unsigned*)&As[r][k] = xsrc[i];
        }
    }
    __syncthreads();

    float s[TOPK]; int id[TOPK];
#pragma unroll
    for (int j = 0; j < TOPK; ++j) { s[j] = -1e30f; id[j] = -1; }

    const int aoff = (lane < 16) ? 0 : 16;  // A frag K sub-offset (contiguous, matches B)
    const int koff = (lane < 16) ? 0 : 16;  // B frag K sub-offset
    const int segBeg = seg * SEGCOLS;
    const int segEnd = segBeg + SEGCOLS;

    for (int tp = wave; tp < NSEGPAIRS; tp += 8) {   // wave-uniform loop
        const int n0 = segBeg + tp * 32;             // local col base in chunk
        // two B columns streams per lane (tiles n0.. and n0+16..)
        const int nloc0 = n0 + ln15;
        const int nloc1 = n0 + 16 + ln15;
        const int ns0 = (nloc0 < CHUNK) ? nloc0 : 0; // guard tail (masked later)
        const int ns1 = (nloc1 < CHUNK) ? nloc1 : 0;
        const unsigned short* erow0 = Ebf + (size_t)ns0 * DDIM + koff;
        const unsigned short* erow1 = Ebf + (size_t)ns1 * DDIM + koff;

        v8f acc0 = {0.f, 0.f, 0.f, 0.f, 0.f, 0.f, 0.f, 0.f};
        v8f acc1 = {0.f, 0.f, 0.f, 0.f, 0.f, 0.f, 0.f, 0.f};
#pragma unroll 4
        for (int k0 = 0; k0 < DDIM; k0 += 32) {
            Frag a, b0, b1;
            a.q[0]  = *(const uint4*)&As[ln15][k0 + aoff];       // K = k0+aoff..+7
            a.q[1]  = *(const uint4*)&As[ln15][k0 + aoff + 8];   // K = +8..+15
            b0.q[0] = *(const uint4*)(erow0 + k0);               // K = k0+koff..+7
            b0.q[1] = *(const uint4*)(erow0 + k0 + 8);           // K = +8..+15
            b1.q[0] = *(const uint4*)(erow1 + k0);
            b1.q[1] = *(const uint4*)(erow1 + k0 + 8);
            acc0 = __builtin_amdgcn_wmma_f32_16x16x32_bf16(
                false, a.v, false, b0.v, (short)0, acc0, false, false);
            acc1 = __builtin_amdgcn_wmma_f32_16x16x32_bf16(
                false, a.v, false, b1.v, (short)0, acc1, false, false);
        }

        // scatter D slab (row = half*8 + r, col = ln15 / 16+ln15)
#pragma unroll
        for (int r = 0; r < 8; ++r) {
            tile[wave][half * 8 + r][ln15]      = acc0[r];
            tile[wave][half * 8 + r][16 + ln15] = acc1[r];
        }

        // scan: lane pair (ln15, half) covers row ln15, cols jb..jb+7 of each tile
        const int jb = half * 8;
#pragma unroll
        for (int j = 0; j < 8; ++j) {
            float v0 = tile[wave][ln15][jb + j];
            int  cl0 = n0 + jb + j;                  // local col within chunk
            if (cl0 < segEnd)                         // mask pair/tile overhang
                insert10(s, id, v0, chunk * CHUNK + cl0);
        }
#pragma unroll
        for (int j = 0; j < 8; ++j) {
            float v1 = tile[wave][ln15][16 + jb + j];
            int  cl1 = n0 + 16 + jb + j;
            if (cl1 < segEnd)
                insert10(s, id, v1, chunk * CHUNK + cl1);
        }
    }

    // publish this lane's list (list id = wave*2 + half) for row ln15
#pragma unroll
    for (int j = 0; j < TOPK; ++j) {
        mrg_s[ln15][wave * 2 + half][j] = s[j];
        mrg_i[ln15][wave * 2 + half][j] = id[j];
    }
    __syncthreads();

    // one thread per row merges 16 lists -> block-level top-10 for this segment
    if (tid < 16) {
        float fs[TOPK]; int fi[TOPK];
#pragma unroll
        for (int j = 0; j < TOPK; ++j) { fs[j] = -1e30f; fi[j] = -1; }
        for (int l = 0; l < 16; ++l)
#pragma unroll
            for (int j = 0; j < TOPK; ++j)
                insert10(fs, fi, mrg_s[tid][l][j], mrg_i[tid][l][j]);

        const int grow = rt * 16 + tid;
        const size_t base = ((size_t)grow * NCANDSEG + chunk * NSEG + seg) * TOPK;
#pragma unroll
        for (int j = 0; j < TOPK; ++j)
            cand[base + j] = make_float2(fs[j], __int_as_float(fi[j]));
    }
}

// ---------------- kernel 2: final merge of 40 segment lists per row ------
__global__ void final_topk_kernel(const float2* __restrict__ cand,
                                  float* __restrict__ out) {
    int row = blockIdx.x * blockDim.x + threadIdx.x;  // 0..4095
    float s[TOPK]; int id[TOPK];
#pragma unroll
    for (int j = 0; j < TOPK; ++j) { s[j] = -1e30f; id[j] = -1; }

    const size_t base = (size_t)row * NCANDSEG * TOPK;
    for (int c = 0; c < NCANDSEG; ++c) {
#pragma unroll
        for (int j = 0; j < TOPK; ++j) {
            float2 p = cand[base + c * TOPK + j];
            insert10(s, id, p.x, __float_as_int(p.y));
        }
    }
#pragma unroll
    for (int j = 0; j < TOPK; ++j) {
        out[row * TOPK + j] = s[j];                              // scores
        out[BROWS * TOPK + row * TOPK + j] = (float)id[j];       // indices
    }
}

// ---------------- host launch ----------------
extern "C" void kernel_launch(void* const* d_in, const int* in_sizes, int n_in,
                              void* d_out, int out_size, void* d_ws, size_t ws_size,
                              hipStream_t stream) {
    const float* x = (const float*)d_in[0];                // [4096, 768]
    const float* E = (const float*)d_in[1];                // [100000, 768]
    // d_in[2] is k == 10 (compile-time TOPK)
    float* out = (float*)d_out;                            // 4096*10 scores, then 4096*10 indices

    char* ws = (char*)d_ws;
    unsigned short* Xbf = (unsigned short*)ws;                       // 6,291,456 B
    unsigned short* Ebf = (unsigned short*)(ws + 6291456);           // 15,360,000 B (one chunk)
    float2*         cand = (float2*)(ws + 6291456 + 15360000);       // 13,107,200 B

    // x -> bf16 once
    cvt_bf16_kernel<<<(BROWS * DDIM / 2) / 256, 256, 0, stream>>>(
        x, Xbf, BROWS * DDIM / 2);

    // stream E chunk-by-chunk: convert to bf16, then fused WMMA GEMM + top-k
    for (int c = 0; c < NCHUNK; ++c) {
        cvt_bf16_kernel<<<(CHUNK * DDIM / 2) / 256, 256, 0, stream>>>(
            E + (size_t)c * CHUNK * DDIM, Ebf, CHUNK * DDIM / 2);
        gemm_topk_kernel<<<dim3(BROWS / 16, NSEG), 256, 0, stream>>>(
            Xbf, Ebf, cand, c);
    }

    // merge per-segment candidates into final top-10 per row
    final_topk_kernel<<<BROWS / 256, 256, 0, stream>>>(cand, out);
}
